// NAOMI_72756745994362
// MI455X (gfx1250) — matrix operand order repaired and checked by reading, not compiled, not run on hardware
//
#include <hip/hip_runtime.h>

// ---------------- types ----------------
typedef __attribute__((ext_vector_type(16))) __bf16 v16bf;
typedef __attribute__((ext_vector_type(8)))  float  v8f;

// ---------------- problem dims ----------------
constexpr int T_N  = 256;   // timesteps
constexpr int B_N  = 512;   // batch
constexpr int D_N  = 64;    // feature dim
constexpr int Hn   = 100;   // rnn dim
constexpr int H3n  = 300;   // 3*H gates
constexpr int BT   = 16;    // batch tile (WMMA M)
constexpr int NTG  = 19;    // n-tiles for 300 -> 304
constexpr int NPG  = 304;   // padded gate width
constexpr int SBLD = 68;    // state row stride (65 -> 68 floats)
constexpr int HBLD = 104;   // h_back row stride (100 -> 104 floats)

// ---------------- workspace layout (bytes) ----------------
constexpr size_t OFF_WBX  = 0;                                  // bgru_W swizzled bf16: 3x19 tiles
constexpr size_t SZ_WBX   = (size_t)3 * 19 * 512 * 2;
constexpr size_t OFF_UB   = OFF_WBX + SZ_WBX;                   // bgru_U: 4x19 tiles
constexpr size_t SZ_UB    = (size_t)4 * 19 * 512 * 2;
constexpr size_t OFF_WF   = OFF_UB + SZ_UB;                     // gru_W: 2x19 tiles
constexpr size_t SZ_WF    = (size_t)2 * 19 * 512 * 2;
constexpr size_t OFF_UF   = OFF_WF + SZ_WF;                     // gru_U: 4x19 tiles
constexpr size_t SZ_UF    = (size_t)4 * 19 * 512 * 2;
constexpr size_t OFF_DEC  = OFF_UF + SZ_UF;                     // dec_W: 4 levels x 7x4 tiles
constexpr size_t SZ_DEC   = (size_t)4 * 7 * 4 * 512 * 2;
constexpr size_t OFF_MEAN = OFF_DEC + SZ_DEC;                   // mean_W: 4 levels x 2x4 tiles
constexpr size_t SZ_MEAN  = (size_t)4 * 2 * 4 * 512 * 2;
constexpr size_t OFF_SBUF = ((OFF_MEAN + SZ_MEAN + 255) / 256) * 256;  // states f32 [T][B][68]
constexpr size_t SZ_SBUF  = (size_t)T_N * B_N * SBLD * 4;
constexpr size_t OFF_GXB  = OFF_SBUF + SZ_SBUF;                 // precomputed x@Wb+b0, bf16 [T][B][304]
constexpr size_t SZ_GXB   = (size_t)T_N * B_N * NPG * 2;
constexpr size_t OFF_HBACK= OFF_GXB + SZ_GXB;                   // h_back_dict f32 [257][B][104]

// ---------------- small helpers ----------------
static __device__ __forceinline__ unsigned short f2bf(float f) {
  unsigned int u = __builtin_bit_cast(unsigned int, f);
  unsigned int r = u + 0x7FFFu + ((u >> 16) & 1u);   // RNE
  return (unsigned short)(r >> 16);
}
static __device__ __forceinline__ float bf2f(unsigned short h) {
  unsigned int u = ((unsigned int)h) << 16;
  return __builtin_bit_cast(float, u);
}
static __device__ __forceinline__ float sigm(float x) {
  return 1.0f / (1.0f + __expf(-x));
}
static __device__ __forceinline__ v16bf ldfrag(const unsigned short* p) {
  v16bf r;
  __builtin_memcpy(&r, p, sizeof(r));
  return r;
}

// Workgroup GEMM: D[16 x 16*NT] += A[16 x 32*KT] * B, bf16 in / f32 out via v_wmma.
// A fragments (LDS, swizzled), B fragments (global ws, pre-swizzled), out to LDS f32.
static __device__ void wg_gemm(const unsigned short* aF, const unsigned short* Bw,
                               int KT, int NT, float* out, int ldo,
                               const float* bias, int nvalid) {
  const int lane = threadIdx.x & 31;
  const int wave = threadIdx.x >> 5;
  for (int nt = wave; nt < NT; nt += 8) {
    v8f acc = {};
    for (int kt = 0; kt < KT; ++kt) {
      v16bf av = ldfrag(aF + ((size_t)kt * 32 + lane) * 16);
      v16bf bv = ldfrag(Bw + (((size_t)kt * NT + nt) * 32 + lane) * 16);
      acc = __builtin_amdgcn_wmma_f32_16x16x32_bf16(false, av, false, bv,
                                                    (short)0, acc, false, false);
    }
    int n = nt * 16 + (lane & 15);
    if (n < nvalid) {
      float bvl = bias[n];
      int mb = (lane >> 4) << 3;
#pragma unroll
      for (int j = 0; j < 8; ++j) out[(size_t)(mb + j) * ldo + n] = acc[j] + bvl;
    }
  }
}

// ---------------- kernel 1: swizzle weights into WMMA B-fragment order (bf16) ----------------
__global__ __launch_bounds__(256) void naomi_prep(
    const float* gruW, const float* gruU, const float* bgruW, const float* bgruU,
    const float* decW, const float* meanW, unsigned char* ws) {
  int bid = blockIdx.x;
  const float* src; int Ks, Ns, NTm; size_t dstOff; int tile;
  if (bid < 57)       { src = bgruW; Ks = 65;  Ns = 300; NTm = 19; dstOff = OFF_WBX / 2; tile = bid; }
  else if (bid < 133) { src = bgruU; Ks = 100; Ns = 300; NTm = 19; dstOff = OFF_UB / 2;  tile = bid - 57; }
  else if (bid < 171) { src = gruW;  Ks = 64;  Ns = 300; NTm = 19; dstOff = OFF_WF / 2;  tile = bid - 133; }
  else if (bid < 247) { src = gruU;  Ks = 100; Ns = 300; NTm = 19; dstOff = OFF_UF / 2;  tile = bid - 171; }
  else if (bid < 359) { int r = bid - 247; int lvl = r / 28; tile = r % 28;
                        src = decW + (size_t)lvl * 200 * 50; Ks = 200; Ns = 50; NTm = 4;
                        dstOff = OFF_DEC / 2 + (size_t)lvl * 7 * 4 * 512; }
  else                { int r = bid - 359; int lvl = r / 8; tile = r % 8;
                        src = meanW + (size_t)lvl * 50 * 64; Ks = 50; Ns = 64; NTm = 4;
                        dstOff = OFF_MEAN / 2 + (size_t)lvl * 2 * 4 * 512; }
  int kt = tile / NTm, nt = tile % NTm;
  (void)nt;
  unsigned short* dst = (unsigned short*)ws + dstOff + (size_t)tile * 512;
  for (int e = threadIdx.x; e < 512; e += 256) {
    int lane = e >> 4, i = e & 15;
    int nn = lane & 15;
    int kk = ((lane >> 4) << 3) + ((i >> 3) << 4) + (i & 7);   // B layout: lanes 0-15 K=0..7,16..23; 16-31 K=8..15,24..31
    int k = kt * 32 + kk, n = (tile % NTm) * 16 + nn;
    float v = (k < Ks && n < Ns) ? src[(size_t)k * Ns + n] : 0.f;
    dst[e] = f2bf(v);
  }
}

// ---------------- kernel 2: build states Sbuf[t][b][0..64] = [raw(64), mask] ----------------
__global__ __launch_bounds__(256) void naomi_sbuf(const float* a, unsigned char* ws) {
  size_t idx = (size_t)blockIdx.x * 256 + threadIdx.x;   // t*(B*SBLD) + b*SBLD + k
  int k = (int)(idx % SBLD);
  int b = (int)((idx / SBLD) % B_N);
  int t = (int)(idx / ((size_t)SBLD * B_N));
  float v;
  if (k < 64)       v = a[(((size_t)b * D_N + k) * T_N + t) * 3 + 0];
  else if (k == 64) v = a[(((size_t)b * D_N + 0) * T_N + t) * 3 + 2];
  else              v = 0.f;
  ((float*)(ws + OFF_SBUF))[idx] = v;
}

// ---------------- kernel 3: parallel precompute Gxb[t] = states[t] @ bgru_W + b0 ----------------
__global__ __launch_bounds__(256) void naomi_gxb(const float* bgru_b, unsigned char* ws) {
  __shared__ alignas(32) unsigned short aF[3 * 512];
  __shared__ float gx[BT][NPG];
  const int tid = threadIdx.x;
  int t  = blockIdx.x >> 5;
  int b0 = (blockIdx.x & 31) * BT;
  const float* Sb = (const float*)(ws + OFF_SBUF) + ((size_t)t * B_N + b0) * SBLD;
  for (int e = tid; e < 3 * 512; e += 256) {           // stage A swizzled (K pad 96)
    int kt = e >> 9, r = e & 511, lane = r >> 4, i = r & 15;
    int m = lane & 15;
    int kk = ((lane >> 4) << 3) + ((i >> 3) << 4) + (i & 7);
    int k = kt * 32 + kk;
    aF[e] = f2bf(k < 65 ? Sb[m * SBLD + k] : 0.f);
  }
  __syncthreads();
  wg_gemm(aF, (const unsigned short*)(ws + OFF_WBX), 3, NTG, &gx[0][0], NPG, bgru_b, H3n);
  __syncthreads();
  unsigned short* Gxb = (unsigned short*)(ws + OFF_GXB) + ((size_t)t * B_N + b0) * NPG;
  for (int e = tid; e < BT * H3n; e += 256) {
    int m = e / H3n, n = e % H3n;
    Gxb[m * NPG + n] = f2bf(gx[m][n]);
  }
}

// ---------------- kernel 4: sequential backward GRU sweep (one WG per batch tile) ----------------
__global__ __launch_bounds__(256) void naomi_bsweep(const float* bgru_b, unsigned char* ws) {
  __shared__ float hf[BT][HBLD];
  __shared__ alignas(32) unsigned short aF[4 * 512];
  __shared__ float gh[BT][NPG];
  const int tid = threadIdx.x;
  const int b0  = blockIdx.x * BT;
  float* Hback = (float*)(ws + OFF_HBACK);
  const unsigned short* Gxb = (const unsigned short*)(ws + OFF_GXB);
  const unsigned short* Ub  = (const unsigned short*)(ws + OFF_UB);
  const float* bb1 = bgru_b + H3n;
  for (int e = tid; e < BT * HBLD; e += 256) hf[e / HBLD][e % HBLD] = 0.f;
  for (size_t o = (size_t)tid * 64; o < SZ_UB; o += 256 * 64)
    __builtin_prefetch(ws + OFF_UB + o, 0, 3);
  __syncthreads();
  for (int t = T_N - 1; t >= 1; --t) {
    // h_back_dict[t+1] = h_back (pre-update)
    for (int e = tid; e < BT * Hn; e += 256) {
      int m = e / Hn, c = e % Hn;
      Hback[((size_t)(t + 1) * B_N + b0 + m) * HBLD + c] = hf[m][c];
    }
    for (int e = tid; e < 4 * 512; e += 256) {         // stage h as A frag (K pad 128)
      int kt = e >> 9, r = e & 511, lane = r >> 4, i = r & 15;
      int m = lane & 15;
      int kk = ((lane >> 4) << 3) + ((i >> 3) << 4) + (i & 7);
      int k = kt * 32 + kk;
      aF[e] = f2bf(k < Hn ? hf[m][k] : 0.f);
    }
    __syncthreads();
    wg_gemm(aF, Ub, 4, NTG, &gh[0][0], NPG, bb1, H3n);
    __syncthreads();
    const unsigned short* gx = Gxb + ((size_t)t * B_N + b0) * NPG;
    for (int e = tid; e < BT * Hn; e += 256) {         // gate math (f32 VALU)
      int m = e / Hn, c = e % Hn;
      float z  = sigm(bf2f(gx[m * NPG + c])          + gh[m][c]);
      float r  = sigm(bf2f(gx[m * NPG + c + Hn])     + gh[m][c + Hn]);
      float nn = tanhf(bf2f(gx[m * NPG + c + 2 * Hn]) + r * gh[m][c + 2 * Hn]);
      hf[m][c] = z * hf[m][c] + (1.f - z) * nn;
    }
    __syncthreads();
  }
}

// ---------------- kernel 5: forward GRU + dyadic interpolation + backward recompute ----------------
__global__ __launch_bounds__(256) void naomi_forward_k(
    const float* a, const float* gru_b, const float* bgru_b,
    const float* dec_b, const float* mean_b, unsigned char* ws) {
  __shared__ float hf[BT][HBLD];
  __shared__ float hb[BT][HBLD];
  __shared__ alignas(32) unsigned short aF[7 * 512];
  __shared__ float gi[BT][NPG];
  __shared__ float gh[BT][NPG];
  __shared__ float dbuf[BT][64];
  __shared__ int obs[T_N];
  const int tid = threadIdx.x;
  const int b0  = blockIdx.x * BT;
  float* Sbuf  = (float*)(ws + OFF_SBUF);
  float* Hback = (float*)(ws + OFF_HBACK);
  const unsigned short* Wf    = (const unsigned short*)(ws + OFF_WF);
  const unsigned short* Uf    = (const unsigned short*)(ws + OFF_UF);
  const unsigned short* Wbx   = (const unsigned short*)(ws + OFF_WBX);
  const unsigned short* Ubx   = (const unsigned short*)(ws + OFF_UB);
  const unsigned short* DecW  = (const unsigned short*)(ws + OFF_DEC);
  const unsigned short* MeanW = (const unsigned short*)(ws + OFF_MEAN);
  const float* gb0 = gru_b;  const float* gb1 = gru_b + H3n;
  const float* bb0 = bgru_b; const float* bb1 = bgru_b + H3n;

  for (int e = tid; e < T_N; e += 256) obs[e] = (a[(size_t)e * 3 + 2] > 0.5f) ? 1 : 0;
  for (int e = tid; e < BT * HBLD; e += 256) hf[e / HBLD][e % HBLD] = 0.f;
  for (size_t o = (size_t)tid * 64; o < SZ_UF; o += 256 * 64)
    __builtin_prefetch(ws + OFF_UF + o, 0, 3);
  __syncthreads();

  auto stageA = [&](int KT, auto&& getv) {
    for (int e = tid; e < KT * 512; e += 256) {
      int kt = e >> 9, r = e & 511, lane = r >> 4, i = r & 15;
      int m = lane & 15;
      int kk = ((lane >> 4) << 3) + ((i >> 3) << 4) + (i & 7);
      aF[e] = f2bf(getv(m, kt * 32 + kk));
    }
    __syncthreads();
  };
  auto gemm = [&](int KT, int NT, const unsigned short* Bw, float* out, int ldo,
                  const float* bias, int nvalid) {
    wg_gemm(aF, Bw, KT, NT, out, ldo, bias, nvalid);
    __syncthreads();
  };
  auto gate = [&](float* hdst) {
    for (int e = tid; e < BT * Hn; e += 256) {
      int m = e / Hn, c = e % Hn;
      float z  = sigm(gi[m][c] + gh[m][c]);
      float r  = sigm(gi[m][c + Hn] + gh[m][c + Hn]);
      float nn = tanhf(gi[m][c + 2 * Hn] + r * gh[m][c + 2 * Hn]);
      float* hp = hdst + m * HBLD + c;
      *hp = z * (*hp) + (1.f - z) * nn;
    }
    __syncthreads();
  };
  auto fwd_step = [&](int t) {
    const float* Sb = Sbuf + ((size_t)t * B_N + b0) * SBLD;
    stageA(2, [&](int m, int k) { return (k < 64) ? Sb[m * SBLD + 1 + k] : 0.f; });
    gemm(2, NTG, Wf, &gi[0][0], NPG, gb0, H3n);
    stageA(4, [&](int m, int k) { return (k < Hn) ? hf[m][k] : 0.f; });
    gemm(4, NTG, Uf, &gh[0][0], NPG, gb1, H3n);
    gate(&hf[0][0]);
  };
  auto load_hb = [&](int key) {
    for (int e = tid; e < BT * Hn; e += 256) {
      int m = e / Hn, c = e % Hn;
      hb[m][c] = Hback[((size_t)key * B_N + b0 + m) * HBLD + c];
    }
    __syncthreads();
  };
  auto store_hb = [&](int key) {
    for (int e = tid; e < BT * Hn; e += 256) {
      int m = e / Hn, c = e % Hn;
      Hback[((size_t)key * B_N + b0 + m) * HBLD + c] = hb[m][c];
    }
    __syncthreads();
  };
  auto bgru_gh_gate = [&]() {
    stageA(4, [&](int m, int k) { return (k < Hn) ? hb[m][k] : 0.f; });
    gemm(4, NTG, Ubx, &gh[0][0], NPG, bb1, H3n);
    gate(&hb[0][0]);
  };

  fwd_step(0);                       // h = gru(states[0][:,1:], 0)
  int curr_p = 0;
  while (curr_p < T_N - 1) {
    if (obs[curr_p + 1]) {
      curr_p += 1;
      fwd_step(curr_p);
    } else {
      int next_p = curr_p + 1;
      while (next_p < T_N && !obs[next_p]) next_p++;
      int step = 1;
      while (curr_p + 2 * step <= next_p && step <= 8) step <<= 1;
      if (step > 1) step >>= 1;
      int lvl  = (step == 8) ? 3 : (step == 4) ? 2 : (step == 2) ? 1 : 0;
      int key  = curr_p + 2 * step;
      int tmid = curr_p + step;
      // ---- decoder: mean = relu([h, hb] @ decW + b) @ meanW + b ----
      load_hb(key);
      stageA(7, [&](int m, int k) {
        return (k < Hn) ? hf[m][k] : ((k < 2 * Hn) ? hb[m][k - Hn] : 0.f);
      });
      gemm(7, 4, DecW + (size_t)lvl * 7 * 4 * 512, &dbuf[0][0], 64, dec_b + lvl * 50, 50);
      for (int e = tid; e < BT * 50; e += 256) {
        int m = e / 50, n = e % 50;
        float v = dbuf[m][n];
        dbuf[m][n] = v > 0.f ? v : 0.f;
      }
      __syncthreads();
      stageA(2, [&](int m, int k) { return (k < 50) ? dbuf[m][k] : 0.f; });
      gemm(2, 4, MeanW + (size_t)lvl * 2 * 4 * 512, &gi[0][0], NPG, mean_b + lvl * 64, 64);
      {
        float* Sb = Sbuf + ((size_t)tmid * B_N + b0) * SBLD;   // new_state -> Sbuf
        for (int e = tid; e < BT * SBLD; e += 256) {
          int m = e / SBLD, k = e % SBLD;
          Sb[m * SBLD + k] = (k < 64) ? gi[m][k] : ((k == 64) ? 1.f : 0.f);
        }
        __threadfence_block();
        __syncthreads();
      }
      if (step > 1) {
        // ---- recompute h_back_dict[right..left] with the interpolated state ----
        int right = curr_p + step, left = curr_p + step / 2;
        load_hb(right + 1);
        // gi for new_state = [mean, 1] (still in gi[0..63] -> stage BEFORE overwrite)
        stageA(3, [&](int m, int k) {
          return (k < 64) ? gi[m][k] : ((k == 64) ? 1.f : 0.f);
        });
        gemm(3, NTG, Wbx, &gi[0][0], NPG, bb0, H3n);
        bgru_gh_gate();
        store_hb(right);
        for (int i2 = right - 1; i2 >= left; --i2) {
          for (int e = tid; e < BT * H3n; e += 256) {          // zeros_in: gi = bias only
            int m = e / H3n, n = e % H3n;
            gi[m][n] = bb0[n];
          }
          __syncthreads();
          bgru_gh_gate();
          store_hb(i2);
        }
      }
      if (tid == 0) obs[tmid] = 1;
      __syncthreads();
    }
  }
}

// ---------------- kernel 6: transpose states -> output [B, D, T, 1] ----------------
__global__ __launch_bounds__(256) void naomi_out(const unsigned char* ws, float* out) {
  size_t idx = (size_t)blockIdx.x * 256 + threadIdx.x;   // (b*D + d)*T + t
  int t = (int)(idx % T_N);
  int d = (int)((idx / T_N) % D_N);
  int b = (int)(idx / ((size_t)T_N * D_N));
  out[idx] = ((const float*)(ws + OFF_SBUF))[((size_t)t * B_N + b) * SBLD + (d + 1)];
}

// ---------------- launcher ----------------
extern "C" void kernel_launch(void* const* d_in, const int* in_sizes, int n_in,
                              void* d_out, int out_size, void* d_ws, size_t ws_size,
                              hipStream_t stream) {
  (void)in_sizes; (void)n_in; (void)out_size; (void)ws_size;
  const float* a      = (const float*)d_in[0];
  const float* gru_W  = (const float*)d_in[1];
  const float* gru_U  = (const float*)d_in[2];
  const float* gru_b  = (const float*)d_in[3];
  const float* bgru_W = (const float*)d_in[4];
  const float* bgru_U = (const float*)d_in[5];
  const float* bgru_b = (const float*)d_in[6];
  const float* dec_W  = (const float*)d_in[7];
  const float* dec_b  = (const float*)d_in[8];
  const float* mean_W = (const float*)d_in[9];
  const float* mean_b = (const float*)d_in[10];
  unsigned char* ws = (unsigned char*)d_ws;
  float* out = (float*)d_out;

  naomi_prep<<<391, 256, 0, stream>>>(gru_W, gru_U, bgru_W, bgru_U, dec_W, mean_W, ws);
  naomi_sbuf<<<(T_N * B_N * SBLD) / 256, 256, 0, stream>>>(a, ws);
  naomi_gxb<<<T_N * 32, 256, 0, stream>>>(bgru_b, ws);
  naomi_bsweep<<<32, 256, 0, stream>>>(bgru_b, ws);
  naomi_forward_k<<<32, 256, 0, stream>>>(a, gru_b, bgru_b, dec_b, mean_b, ws);
  naomi_out<<<(B_N * D_N * T_N) / 256, 256, 0, stream>>>(ws, out);
}
